// HyperbolicMLR_16114717294899
// MI455X (gfx1250) — matrix operand order repaired and checked by reading, not compiled, not run on hardware
//
#include <hip/hip_runtime.h>

// ---------------------------------------------------------------------------
// Hyperbolic MLR for MI455X (gfx1250, wave32).
//   out[b,c] = -2*atanh(min(||mobius_add(-x_b, proto_c)||, 1-1e-5)) * exp(ls_c)
// GEMM xy = x @ protos^T via V_WMMA_F32_16X16X32_BF16 with a bf16 hi+lo split
// (3 WMMAs per tile-K: hh + hl + lh) recovering ~fp32 product accuracy.
// Operands pre-packed into the exact CDNA5 WMMA fragment layouts; the GEMM
// streams fragments straight from L2 with 32B vector loads (no LDS).
// Fully software-pipelined: B frags double-buffered across K-steps; A slot
// refills are issued immediately after the WMMAs that release each slot, so
// every load has >=6 WMMAs of latency cover.
// ---------------------------------------------------------------------------

typedef __attribute__((ext_vector_type(16))) __bf16 v16bf;
typedef __attribute__((ext_vector_type(8)))  float  v8f;

#define D_DIM   256
#define KSTEPS  (D_DIM / 32)          // 8 K-steps of 32
#define MAXNORM 0.99999f              // 1/sqrt(c) - 1e-5, c = 1

// ---- fragment-layout index helpers (ISA 7.12.2, wave32) -------------------
// A 16x32 bf16: lane = m + 16*h ; e<8 -> K = 8h+e ; e>=8 -> K = 16+8h+(e-8)
__device__ __forceinline__ size_t a_frag_idx(int row, int k) {
  int bt = row >> 4, m = row & 15;
  int ks = k >> 5, kk = k & 31;
  int hi16 = kk & 16;                 // 0 or 16
  int rem  = kk & 15;
  int h    = rem >> 3;                // lane half
  int e    = (hi16 ? 8 : 0) + (rem & 7);
  int lane = m + (h << 4);
  return ((size_t)(bt * KSTEPS + ks) * 32 + lane) * 16 + e;
}
// B 32x16 bf16: lane = n + 16*h ; e<8 -> K = 8h+e ; e>=8 -> K = 16+8h+(e-8)
__device__ __forceinline__ size_t b_frag_idx(int col, int k) {
  int ct = col >> 4, n = col & 15;
  int ks = k >> 5, kk = k & 31;
  int lane = n + (((kk & 15) >> 3) << 4);
  int e    = ((kk & 16) ? 8 : 0) + (kk & 7);
  return ((size_t)(ct * KSTEPS + ks) * 32 + lane) * 16 + e;
}

__device__ __forceinline__ float wave_sum32(float v) {
#pragma unroll
  for (int off = 16; off > 0; off >>= 1) v += __shfl_xor(v, off, 32);
  return v;
}

__device__ __forceinline__ v8f wmma_bf16(v16bf a, v16bf b, v8f c) {
  return __builtin_amdgcn_wmma_f32_16x16x32_bf16(false, a, false, b,
                                                 (short)0, c, false, false);
}

// ---- prep kernel: x -> (x2, A-fragment hi/lo bf16) ------------------------
__global__ __launch_bounds__(256) void hmlr_prep_x(
    const float* __restrict__ x, __bf16* __restrict__ a_hi,
    __bf16* __restrict__ a_lo, float* __restrict__ x2, int Brows) {
  const int lane = threadIdx.x & 31, wid = threadIdx.x >> 5;
  const int row = blockIdx.x * 8 + wid;
  if (row >= Brows) return;
  const float* xr = x + (size_t)row * D_DIM + lane * 8;
  float v[8];
#pragma unroll
  for (int j = 0; j < 8; ++j) v[j] = xr[j];
  float s = 0.f;
#pragma unroll
  for (int j = 0; j < 8; ++j) s += v[j] * v[j];
  s = wave_sum32(s);
  if (lane == 0) x2[row] = s;
#pragma unroll
  for (int j = 0; j < 8; ++j) {
    int k = lane * 8 + j;
    size_t idx = a_frag_idx(row, k);
    __bf16 h = (__bf16)v[j];
    a_hi[idx] = h;
    a_lo[idx] = (__bf16)(v[j] - (float)h);
  }
}

// ---- prep kernel: proto_tangent -> expmap0+project -> B-fragments ---------
__global__ __launch_bounds__(256) void hmlr_prep_protos(
    const float* __restrict__ pt, const float* __restrict__ ls,
    __bf16* __restrict__ b_hi, __bf16* __restrict__ b_lo,
    float* __restrict__ y2, float* __restrict__ sc, int Ccols) {
  const int lane = threadIdx.x & 31, wid = threadIdx.x >> 5;
  const int row = blockIdx.x * 8 + wid;
  if (row >= Ccols) return;
  const float* vr = pt + (size_t)row * D_DIM + lane * 8;
  float v[8];
#pragma unroll
  for (int j = 0; j < 8; ++j) v[j] = vr[j];
  float s = 0.f;
#pragma unroll
  for (int j = 0; j < 8; ++j) s += v[j] * v[j];
  s = wave_sum32(s);
  float vn = fmaxf(sqrtf(s), 1e-15f);
  float t  = tanhf(vn);                      // == ||exp_0(v)||
  float fac = t / vn;                        // exp map scale
  if (t > MAXNORM) fac *= MAXNORM / t;       // ball projection
  float pn = fminf(t, MAXNORM);
  if (lane == 0) {
    y2[row] = pn * pn;
    sc[row] = __expf(ls[row]);
  }
#pragma unroll
  for (int j = 0; j < 8; ++j) {
    int k = lane * 8 + j;
    size_t idx = b_frag_idx(row, k);
    float p = v[j] * fac;
    __bf16 h = (__bf16)p;
    b_hi[idx] = h;
    b_lo[idx] = (__bf16)(p - (float)h);
  }
}

// ---- main GEMM + fused hyperbolic-distance epilogue -----------------------
// Block = 256 threads = 8 waves, tile 128(M) x 128(N). Waves 2x4; each wave
// owns 4x2 tiles of 16x16 (64 rows x 32 cols).
__global__ __launch_bounds__(256) void hmlr_gemm(
    const __bf16* __restrict__ a_hi, const __bf16* __restrict__ a_lo,
    const __bf16* __restrict__ b_hi, const __bf16* __restrict__ b_lo,
    const float* __restrict__ x2, const float* __restrict__ y2,
    const float* __restrict__ sc, float* __restrict__ out, int Ccols) {
  const int lane   = threadIdx.x & 31;
  const int wid    = threadIdx.x >> 5;
  const int wave_m = wid >> 2;                     // 0..1
  const int wave_n = wid & 3;                      // 0..3
  const int bt0 = blockIdx.y * 8 + wave_m * 4;     // first of 4 M-tiles
  const int ct0 = blockIdx.x * 8 + wave_n * 2;     // first of 2 N-tiles

  v8f acc[4][2];
#pragma unroll
  for (int i = 0; i < 4; ++i)
#pragma unroll
    for (int j = 0; j < 2; ++j)
      acc[i][j] = (v8f){0.f,0.f,0.f,0.f,0.f,0.f,0.f,0.f};

  // fragment buffers: B double-buffered over K-steps, A ping-ponged over M
  v16bf bh0[2], bl0[2], bh1[2], bl1[2];
  v16bf ah[2], al[2];

#define LOAD_B(buf, ks_)                                                      \
  {                                                                           \
    size_t o0 = ((size_t)((ct0 + 0) * KSTEPS + (ks_)) * 32 + lane) * 16;      \
    size_t o1 = ((size_t)((ct0 + 1) * KSTEPS + (ks_)) * 32 + lane) * 16;      \
    bh##buf[0] = *(const v16bf*)(b_hi + o0);                                  \
    bl##buf[0] = *(const v16bf*)(b_lo + o0);                                  \
    bh##buf[1] = *(const v16bf*)(b_hi + o1);                                  \
    bl##buf[1] = *(const v16bf*)(b_lo + o1);                                  \
  }

#define LOAD_A(slot, i_, ks_)                                                 \
  {                                                                           \
    size_t oa = ((size_t)((bt0 + (i_)) * KSTEPS + (ks_)) * 32 + lane) * 16;   \
    ah[slot] = *(const v16bf*)(a_hi + oa);                                    \
    al[slot] = *(const v16bf*)(a_lo + oa);                                    \
  }

  // hi*hi + hi*lo + lo*hi  (lo*lo is below fp32 ulp; dropped)
#define TILE_WMMA(slot, buf, i_)                                              \
  acc[i_][0] = wmma_bf16(ah[slot], bh##buf[0], acc[i_][0]);                   \
  acc[i_][0] = wmma_bf16(ah[slot], bl##buf[0], acc[i_][0]);                   \
  acc[i_][0] = wmma_bf16(al[slot], bh##buf[0], acc[i_][0]);                   \
  acc[i_][1] = wmma_bf16(ah[slot], bh##buf[1], acc[i_][1]);                   \
  acc[i_][1] = wmma_bf16(ah[slot], bl##buf[1], acc[i_][1]);                   \
  acc[i_][1] = wmma_bf16(al[slot], bh##buf[1], acc[i_][1]);

  // Slot refills are issued right AFTER the WMMAs that release the slot, so
  // each A load has >= 6 WMMAs of latency cover. Tiles 0/1 of step ksn_ are
  // refilled in the tail of step ks_ (the wrap on the last step loads step 0
  // again; 4 wasted b128 loads once, keeps the loop branch-free).
#define COMPUTE(buf, ks_, ksn_)                                               \
  {                                                                           \
    TILE_WMMA(0, buf, 0);                                                     \
    LOAD_A(0, 2, ks_);                                                        \
    TILE_WMMA(1, buf, 1);                                                     \
    LOAD_A(1, 3, ks_);                                                        \
    TILE_WMMA(0, buf, 2);                                                     \
    LOAD_A(0, 0, ksn_);                                                       \
    TILE_WMMA(1, buf, 3);                                                     \
    LOAD_A(1, 1, ksn_);                                                       \
  }

  LOAD_B(0, 0);
  LOAD_A(0, 0, 0);                   // peeled preload of first two A tiles
  LOAD_A(1, 1, 0);
#pragma unroll 1
  for (int ks = 0; ks < KSTEPS; ks += 2) {
    LOAD_B(1, ks + 1);               // next K-step's B in flight during compute
    COMPUTE(0, ks, ks + 1);
    LOAD_B(0, (ks + 2) & (KSTEPS - 1));
    COMPUTE(1, ks + 1, (ks + 2) & (KSTEPS - 1));
  }
#undef LOAD_B
#undef LOAD_A
#undef TILE_WMMA
#undef COMPUTE

  // Fused epilogue. C/D layout: element r of acc -> row = 16*tile + 8*(lane/16)+r,
  // col = 16*tile + (lane%16).
  const int ncol  = lane & 15;
  const int mhalf = (lane >> 4) << 3;
#pragma unroll
  for (int i = 0; i < 4; ++i) {
    const int row_base = (bt0 + i) * 16 + mhalf;
    float x2v[8];
#pragma unroll
    for (int r = 0; r < 8; ++r) x2v[r] = x2[row_base + r];
#pragma unroll
    for (int j = 0; j < 2; ++j) {
      const int col = (ct0 + j) * 16 + ncol;
      const float y2v = y2[col];
      const float scv = sc[col];
#pragma unroll
      for (int r = 0; r < 8; ++r) {
        const float nxy   = -acc[i][j][r];               // <-x, y>
        const float av    = 1.f + 2.f * nxy + y2v;       // coeff of -x
        const float bv    = 1.f - x2v[r];                // coeff of y
        const float denom = fmaxf(1.f + 2.f * nxy + x2v[r] * y2v, 1e-15f);
        const float numsq = av * av * x2v[r] + 2.f * av * bv * nxy + bv * bv * y2v;
        float nrm = sqrtf(fmaxf(numsq, 0.f)) / denom;
        nrm = fminf(nrm, MAXNORM);
        const float dist = __logf((1.f + nrm) / (1.f - nrm));   // 2*atanh
        __builtin_nontemporal_store(-dist * scv,
            out + (size_t)(row_base + r) * Ccols + col);
      }
    }
  }
}

// ---------------------------------------------------------------------------
extern "C" void kernel_launch(void* const* d_in, const int* in_sizes, int n_in,
                              void* d_out, int out_size, void* d_ws, size_t ws_size,
                              hipStream_t stream) {
  const float* x  = (const float*)d_in[0];
  const float* pt = (const float*)d_in[1];
  const float* ls = (const float*)d_in[2];
  float* out = (float*)d_out;

  const int Brows = in_sizes[0] / D_DIM;   // 4096
  const int Ccols = in_sizes[2];           // 8192

  // workspace carve-up (~12.1 MB)
  __bf16* a_hi = (__bf16*)d_ws;
  __bf16* a_lo = a_hi + (size_t)Brows * D_DIM;
  __bf16* b_hi = a_lo + (size_t)Brows * D_DIM;
  __bf16* b_lo = b_hi + (size_t)Ccols * D_DIM;
  float*  x2   = (float*)(b_lo + (size_t)Ccols * D_DIM);
  float*  y2   = x2 + Brows;
  float*  sc   = y2 + Ccols;

  hmlr_prep_x<<<(Brows + 7) / 8, 256, 0, stream>>>(x, a_hi, a_lo, x2, Brows);
  hmlr_prep_protos<<<(Ccols + 7) / 8, 256, 0, stream>>>(pt, ls, b_hi, b_lo, y2, sc, Ccols);

  dim3 grid(Ccols / 128, Brows / 128);     // 64 x 32 blocks
  hmlr_gemm<<<grid, 256, 0, stream>>>(a_hi, a_lo, b_hi, b_lo, x2, y2, sc, out, Ccols);
}
